// QKVAttention_73272142070173
// MI455X (gfx1250) — matrix operand order repaired and checked by reading, not compiled, not run on hardware
//
#include <hip/hip_runtime.h>
#include <hip/hip_bf16.h>
#include <cstddef>

// ---------------------------------------------------------------------------
// QKV cross-head attention for MI455X (gfx1250, wave32, WMMA).
//   score[b,i,j,s,t] = Q[b,i,s,:].K[b,j,t,:] / 8 ; softmax over t
//   ctx[b,i,s,:]     = sum_j W[b,i,j,s,:] @ V[b,j]
// Outputs (concatenated in d_out): context [B,S,H*D] f32, weights [B,H,H,S,S] f32.
// valid_lens is an input but is UNUSED by the reference -> ignored here.
// GEMMs in f16 via v_wmma_f32_16x16x32_f16; softmax/accum in f32.
// Weights/ctx stores are non-temporal: streamed 537 MB output must not evict
// the highly-reused K/V (8 MB, read by 256 WGs each) from the 192 MB L2.
// ---------------------------------------------------------------------------

typedef _Float16 v16h __attribute__((ext_vector_type(16)));
typedef _Float16 v8h  __attribute__((ext_vector_type(8)));
typedef _Float16 v4h  __attribute__((ext_vector_type(4)));
typedef float    v8f  __attribute__((ext_vector_type(8)));
typedef float    v4f  __attribute__((ext_vector_type(4)));

#define B_   2
#define H_   8
#define S_   1024
#define D_   64
#define BM   32          // query rows per workgroup
#define CH   128         // key/value chunk (t dimension)
#define KSTR 72          // LDS stride (halfs) for K chunk [CH][D] (+pad)
#define VSTR 136         // LDS stride (halfs) for V^T chunk [D][CH] (+pad)
#define SSTR 1028        // LDS stride (floats) for score strip [BM][S] (+pad)
#define CTX_ELEMS ((size_t)B_ * S_ * H_ * D_)   // 1,048,576

static __device__ inline v16h cat8(v8h lo, v8h hi) {
  return __builtin_shufflevector(lo, hi, 0,1,2,3,4,5,6,7,8,9,10,11,12,13,14,15);
}

// A fragment (16x32 f16, M x K) per CDNA5 layout:
// lanes 0-15: row M=lane, halves[0..7]=K(k0..k0+7), halves[8..15]=K(k0+16..k0+23)
// lanes16-31: row M=lane-16, K(k0+8..15) and K(k0+24..31). Two ds_load_b128.
static __device__ inline v16h loadA_f16(const _Float16* buf, int stride,
                                        int row0, int k0, int lane) {
  int m   = row0 + (lane & 15);
  int sel = (lane >> 4) & 1;
  const _Float16* p = buf + m * stride + k0 + sel * 8;
  return cat8(*(const v8h*)p, *(const v8h*)(p + 16));
}

// B fragment (32x16 f16, K x N): lane n<16 holds B[k0..k0+7, k0+16..k0+23][n],
// lanes 16-31 hold B[k0+8..k0+15, k0+24..k0+31][n-16] (matches A layout)
// when buf is laid out with N-major rows (stride) and K contiguous.
static __device__ inline v16h loadB_f16(const _Float16* buf, int stride,
                                        int k0, int col0, int lane) {
  int n = col0 + (lane & 15);
  const _Float16* p = buf + n * stride + k0 + ((lane >> 4) & 1) * 8;
  return cat8(*(const v8h*)p, *(const v8h*)(p + 16));
}

// A fragment sourced from f32 LDS (softmax weights) with on-the-fly f16 cvt.
static __device__ inline v16h loadA_f32(const float* buf, int stride,
                                        int row0, int k0, int lane) {
  int m = row0 + (lane & 15);
  const float* p = buf + m * stride + k0 + ((lane >> 4) & 1) * 8;
  v4f a0 = *(const v4f*)p;
  v4f a1 = *(const v4f*)(p + 4);
  v4f a2 = *(const v4f*)(p + 16);
  v4f a3 = *(const v4f*)(p + 20);
  v16h a;
#pragma unroll
  for (int e = 0; e < 4; ++e) {
    a[e]      = (_Float16)a0[e];
    a[4 + e]  = (_Float16)a1[e];
    a[8 + e]  = (_Float16)a2[e];
    a[12 + e] = (_Float16)a3[e];
  }
  return a;
}

static __device__ inline v8f wmma16(v16h a, v16h b, v8f c) {
  return __builtin_amdgcn_wmma_f32_16x16x32_f16(false, a, false, b,
                                                (short)0, c, false, false);
}

__global__ __launch_bounds__(256)
void qkv_attn_gfx1250(const float* __restrict__ Q, const float* __restrict__ K,
                      const float* __restrict__ V, float* __restrict__ out) {
  __shared__ _Float16 sQ[BM * KSTR];    //  4,608 B : Q tile, f16
  __shared__ _Float16 sKV[CH * KSTR];   // 18,432 B : K chunk [t][d] or V^T [d][t]
  __shared__ float    sS[BM * SSTR];    // 131,584 B: score/weight strip [BM][S]
                                        // total ~154.6 KB -> 2 WGs / 320KB WGP

  const int tid  = threadIdx.x;
  const int wave = tid >> 5;
  const int lane = tid & 31;
  const int s0   = blockIdx.x * BM;     // query-tile start
  const int head = blockIdx.y;          // i (query head)
  const int b    = blockIdx.z;

  // ---- stage Q tile: f32 global -> f16 LDS (row-major [m][d]) ----
  {
    const v4f* Qg = (const v4f*)(Q + (((size_t)b * H_ + head) * S_ + s0) * D_);
#pragma unroll
    for (int it = 0; it < (BM * D_ / 4) / 256; ++it) {
      int f = tid + it * 256;
      v4f v = Qg[f];
      int t = f >> 4, dblk = (f & 15) * 4;
      v4h h;
#pragma unroll
      for (int e = 0; e < 4; ++e) h[e] = (_Float16)v[e];
      *(v4h*)(sQ + t * KSTR + dblk) = h;
    }
  }
  __syncthreads();

  // phase-1 wave tiling: rowfrag rf = wave&1 (16 rows), colfrag pair = wave>>1
  const int rf = wave & 1;
  const v16h aQ0 = loadA_f16(sQ, KSTR, rf * 16, 0, lane);   // K=0..31
  const v16h aQ1 = loadA_f16(sQ, KSTR, rf * 16, 32, lane);  // K=32..63

  // phase-2 wave tiling: ctx is 32x64 -> 2x4 fragments, one per wave
  const int mf = wave & 1, nf = wave >> 1;
  v8f cacc = {};

  float* wout = out + CTX_ELEMS;

  for (int j = 0; j < H_; ++j) {
    const float* Kj = K + (((size_t)b * H_ + j) * S_) * D_;
    const float* Vj = V + (((size_t)b * H_ + j) * S_) * D_;

    // ================= phase 1: scores = Q.K^T / 8 =================
    for (int tc = 0; tc < S_ / CH; ++tc) {
      if (tc + 1 < S_ / CH)
        __builtin_prefetch(Kj + (size_t)(tc + 1) * CH * D_ + tid * 4, 0, 1);
      const v4f* Kg = (const v4f*)(Kj + (size_t)tc * CH * D_);
#pragma unroll
      for (int it = 0; it < (CH * D_ / 4) / 256; ++it) {
        int f = tid + it * 256;
        v4f v = Kg[f];
        int t = f >> 4, dblk = (f & 15) * 4;
        v4h h;
#pragma unroll
        for (int e = 0; e < 4; ++e) h[e] = (_Float16)v[e];
        *(v4h*)(sKV + t * KSTR + dblk) = h;
      }
      __syncthreads();

      // Preload ALL four B fragments first (one DS clause), then issue the
      // four WMMAs back-to-back to avoid per-WMMA s_wait_dscnt stalls.
      const int col0 = ((wave >> 1) * 2) * 16;       // this wave's 32 key cols
      v16h b00 = loadB_f16(sKV, KSTR, 0,  col0,      lane);
      v16h b01 = loadB_f16(sKV, KSTR, 32, col0,      lane);
      v16h b10 = loadB_f16(sKV, KSTR, 0,  col0 + 16, lane);
      v16h b11 = loadB_f16(sKV, KSTR, 32, col0 + 16, lane);
      v8f acc0 = {}, acc1 = {};
      acc0 = wmma16(aQ0, b00, acc0);
      acc1 = wmma16(aQ0, b10, acc1);
      acc0 = wmma16(aQ1, b01, acc0);
      acc1 = wmma16(aQ1, b11, acc1);

      // scatter C (f32 16x16: VGPR r -> M=r or r+8; N=lane&15) scaled by 1/8
      {
        int n0    = tc * CH + col0 + (lane & 15);
        int mbase = rf * 16 + ((lane >> 4) & 1) * 8;
#pragma unroll
        for (int r = 0; r < 8; ++r) {
          sS[(mbase + r) * SSTR + n0]      = acc0[r] * 0.125f;
          sS[(mbase + r) * SSTR + n0 + 16] = acc1[r] * 0.125f;
        }
      }
      __syncthreads();
    }

    // ================= softmax over t (len S) per row ==============
    {
      const int r = tid >> 3;            // row 0..31
      const int q = tid & 7;             // 8-thread team per row (same wave)
      float* row = sS + r * SSTR + q * (S_ / 8);
      float mx = -3.402823e38f;
      for (int c = 0; c < S_ / 8; c += 4) {
        v4f v = *(v4f*)(row + c);
        mx = fmaxf(mx, fmaxf(fmaxf(v.x, v.y), fmaxf(v.z, v.w)));
      }
      mx = fmaxf(mx, __shfl_xor(mx, 1));
      mx = fmaxf(mx, __shfl_xor(mx, 2));
      mx = fmaxf(mx, __shfl_xor(mx, 4));
      float sum = 0.f;
      for (int c = 0; c < S_ / 8; c += 4) {
        v4f v = *(v4f*)(row + c);
        v.x = __expf(v.x - mx); v.y = __expf(v.y - mx);
        v.z = __expf(v.z - mx); v.w = __expf(v.w - mx);
        *(v4f*)(row + c) = v;
        sum += (v.x + v.y) + (v.z + v.w);
      }
      sum += __shfl_xor(sum, 1);
      sum += __shfl_xor(sum, 2);
      sum += __shfl_xor(sum, 4);
      float inv = 1.f / sum;
      // normalized weights: non-temporal write to global + back to LDS for W@V
      v4f* wg = (v4f*)(wout +
                ((((size_t)(b * H_ + head)) * H_ + j) * S_ + (s0 + r)) * S_ +
                q * (S_ / 8));
      for (int c = 0; c < S_ / 8; c += 4) {
        v4f v = *(v4f*)(row + c);
        v = v * inv;
        *(v4f*)(row + c) = v;
        __builtin_nontemporal_store(v, wg + (c >> 2));
      }
    }
    __syncthreads();

    // ================= phase 2: ctx += W @ V =======================
    for (int tc = 0; tc < S_ / CH; ++tc) {
      if (tc + 1 < S_ / CH)
        __builtin_prefetch(Vj + (size_t)(tc + 1) * CH * D_ + tid * 4, 0, 1);
      const v4f* Vg = (const v4f*)(Vj + (size_t)tc * CH * D_);
#pragma unroll
      for (int it = 0; it < (CH * D_ / 4) / 256; ++it) {
        int f = tid + it * 256;
        v4f v = Vg[f];
        int t = f >> 4, dblk = (f & 15) * 4;   // store transposed: sKV[d][t]
        sKV[(dblk + 0) * VSTR + t] = (_Float16)v.x;
        sKV[(dblk + 1) * VSTR + t] = (_Float16)v.y;
        sKV[(dblk + 2) * VSTR + t] = (_Float16)v.z;
        sKV[(dblk + 3) * VSTR + t] = (_Float16)v.w;
      }
      __syncthreads();

      // Preload the four V B-fragments as one DS clause, then chain WMMAs;
      // the f32->f16 weight-fragment builds overlap with the previous WMMA.
      v16h bV0 = loadB_f16(sKV, VSTR, 0,  nf * 16, lane);
      v16h bV1 = loadB_f16(sKV, VSTR, 32, nf * 16, lane);
      v16h bV2 = loadB_f16(sKV, VSTR, 64, nf * 16, lane);
      v16h bV3 = loadB_f16(sKV, VSTR, 96, nf * 16, lane);
      cacc = wmma16(loadA_f32(sS, SSTR, mf * 16, tc * CH + 0,  lane), bV0, cacc);
      cacc = wmma16(loadA_f32(sS, SSTR, mf * 16, tc * CH + 32, lane), bV1, cacc);
      cacc = wmma16(loadA_f32(sS, SSTR, mf * 16, tc * CH + 64, lane), bV2, cacc);
      cacc = wmma16(loadA_f32(sS, SSTR, mf * 16, tc * CH + 96, lane), bV3, cacc);
      __syncthreads();
    }
  }

  // ---- write context: out[b, s0+m, head*D + d] (streamed, non-temporal) ----
  {
    int n     = nf * 16 + (lane & 15);
    int mbase = mf * 16 + ((lane >> 4) & 1) * 8;
    float* cg = out + ((size_t)b * S_ + s0) * (H_ * D_) + head * D_ + n;
#pragma unroll
    for (int r = 0; r < 8; ++r)
      __builtin_nontemporal_store(cacc[r], cg + (size_t)(mbase + r) * (H_ * D_));
  }
}

extern "C" void kernel_launch(void* const* d_in, const int* in_sizes, int n_in,
                              void* d_out, int out_size, void* d_ws, size_t ws_size,
                              hipStream_t stream) {
  (void)in_sizes; (void)n_in; (void)out_size; (void)d_ws; (void)ws_size;
  const float* Q = (const float*)d_in[0];
  const float* K = (const float*)d_in[1];
  const float* V = (const float*)d_in[2];
  // d_in[3] (valid_lens) is unused by the reference computation.
  float* out = (float*)d_out;
  dim3 grid(S_ / BM, H_, B_);   // 32 x 8 x 2 = 512 workgroups
  qkv_attn_gfx1250<<<grid, dim3(256), 0, stream>>>(Q, K, V, out);
}
